// TripletLoss_83133386982172
// MI455X (gfx1250) — matrix operand order repaired and checked by reading, not compiled, not run on hardware
//
#include <hip/hip_runtime.h>
#include <stdint.h>

typedef __attribute__((ext_vector_type(16))) _Float16 v16h;
typedef __attribute__((ext_vector_type(8)))  _Float16 v8h;
typedef __attribute__((ext_vector_type(8)))  float    v8f;

#define N_FEATS 4096
#define DIM     512
#define KSTEPS  (DIM / 32)        // 16 WMMA K-chunks of 32
#define PANEL_COLS 32             // columns staged per LDS panel (2 x 16-col tiles)
#define NPANELS (N_FEATS / PANEL_COLS)   // 128 panel iterations
#define PANEL_BYTES (PANEL_COLS * DIM * 2)   // 32 KB
#define MARGIN  0.2f
#define EPS     1e-6f
#define BIG     1e9f

// ---------------------------------------------------------------- kernel 0
__global__ __launch_bounds__(256) void cvt_f32_to_f16(const float* __restrict__ src,
                                                      _Float16* __restrict__ dst,
                                                      int n) {
    int i = blockIdx.x * 256 + threadIdx.x;
    if (i < n) dst[i] = (_Float16)src[i];
}

// ---------------------------------------------------------------- kernel 1
// 8 waves per block, one 16-anchor row tile per wave. The block cooperatively
// double-buffers 32-column B panels in LDS via global_load_async_to_lds_b128
// (ASYNCcnt). Each panel is consumed with a fused K-loop over both 16-column
// sub-tiles (dual accumulators) so every ds-load wait covers two WMMAs.
__global__ __launch_bounds__(256) void triplet_mine(const _Float16* __restrict__ F,
                                                    const int* __restrict__ labels,
                                                    float* __restrict__ hpOut,
                                                    float* __restrict__ hnOut) {
    __shared__ __align__(128) _Float16 panel[2][PANEL_COLS * DIM];   // 2 x 32 KB

    const int lane = threadIdx.x & 31;
    const int wave = threadIdx.x >> 5;
    const int rowTile = blockIdx.x * 8 + wave;   // 32 blocks * 8 waves = 256 tiles
    const int i0 = rowTile * 16;
    const int l15 = lane & 15;
    const int hi  = lane >> 4;                   // which 16-lane half

    // ---- A tile (full K=512) in registers; contiguous 16-bit A layout:
    // lanes 0-15 hold row M=lane, K {0..15}+k0; lanes 16-31 K {16..31}+k0.
    const _Float16* arow = F + (size_t)(i0 + l15) * DIM + hi * 16;
    v16h A[KSTEPS];
#pragma unroll
    for (int ks = 0; ks < KSTEPS; ++ks) {
        v8h p0 = *(const v8h*)(arow + ks * 32);
        v8h p1 = *(const v8h*)(arow + ks * 32 + 8);
        v16h a;
#pragma unroll
        for (int t = 0; t < 8; ++t) { a[t] = p0[t]; a[t + 8] = p1[t]; }
        A[ks] = a;
    }

    // C-slot (v, lane) -> anchor M = v + 8*hi (ISA f32 C/D layout)
    int labI[8];
#pragma unroll
    for (int v = 0; v < 8; ++v) labI[v] = labels[i0 + v + 8 * hi];

    float hp[8], hn[8];
#pragma unroll
    for (int v = 0; v < 8; ++v) { hp[v] = -BIG; hn[v] = BIG; }

    // Async-copy plumbing: panel rows are contiguous in F (32 rows * 1 KB).
    // Wave w copies bytes [w*4096, w*4096+4096): 8 x 512B b128 bursts.
    const uint32_t ldsBuf0  = (uint32_t)(uintptr_t)&panel[0][0];
    const uint32_t ldsBuf1  = (uint32_t)(uintptr_t)&panel[1][0];
    const uint32_t copyOff  = (uint32_t)wave * 4096u + (uint32_t)lane * 16u;
    const uint64_t gBase    = (uint64_t)(uintptr_t)F;

    // prologue: stage panel 0 into buffer 0
    {
        uint64_t g = gBase + copyOff;
        uint32_t l = ldsBuf0 + copyOff;
#pragma unroll
        for (int q = 0; q < 8; ++q) {
            asm volatile("global_load_async_to_lds_b128 %0, %1, off"
                         :: "v"(l + (uint32_t)(q * 512)), "v"(g + (uint64_t)(q * 512))
                         : "memory");
        }
    }

    for (int pt = 0; pt < NPANELS; ++pt) {
        const int cur = pt & 1;

        if (pt + 1 < NPANELS) {
            // kick off panel pt+1 into the other buffer, then wait for panel pt
            // (leave the 8 just-issued copies outstanding: ASYNCcnt <= 8).
            uint64_t g = gBase + (uint64_t)(pt + 1) * PANEL_BYTES + copyOff;
            uint32_t l = (cur ? ldsBuf0 : ldsBuf1) + copyOff;
#pragma unroll
            for (int q = 0; q < 8; ++q) {
                asm volatile("global_load_async_to_lds_b128 %0, %1, off"
                             :: "v"(l + (uint32_t)(q * 512)), "v"(g + (uint64_t)(q * 512))
                             : "memory");
            }
            asm volatile("s_wait_asynccnt 0x8" ::: "memory");
        } else {
            asm volatile("s_wait_asynccnt 0x0" ::: "memory");
        }
        __syncthreads();   // panel pt visible to all waves

        // Fused K-loop over both 16-column sub-tiles, dual accumulators.
        // B fragment: column N=l15 is panel row (sub*16+l15);
        // lanes 0-15 read K=k0..k0+15, lanes 16-31 K=k0+16..k0+31 (32B/lane).
        const _Float16* brow0 = &panel[cur][l15 * DIM + hi * 16];
        const _Float16* brow1 = &panel[cur][(16 + l15) * DIM + hi * 16];
        v8f c0 = {}, c1 = {};
#pragma unroll
        for (int ks = 0; ks < KSTEPS; ++ks) {
            v16h b0 = *(const v16h*)(brow0 + ks * 32);
            v16h b1 = *(const v16h*)(brow1 + ks * 32);
            c0 = __builtin_amdgcn_wmma_f32_16x16x32_f16(
                    false, A[ks], false, b0, (short)0, c0, false, false);
            c1 = __builtin_amdgcn_wmma_f32_16x16x32_f16(
                    false, A[ks], false, b1, (short)0, c1, false, false);
        }

        const int labJ0 = labels[pt * PANEL_COLS + l15];
        const int labJ1 = labels[pt * PANEL_COLS + 16 + l15];
#pragma unroll
        for (int v = 0; v < 8; ++v) {
            float d0 = 1.0f - c0[v];
            if (labI[v] == labJ0) {
                if (d0 > EPS && d0 > hp[v]) hp[v] = d0;   // hardest positive (max dist)
            } else {
                if (d0 < hn[v]) hn[v] = d0;               // hardest negative (min dist)
            }
            float d1 = 1.0f - c1[v];
            if (labI[v] == labJ1) {
                if (d1 > EPS && d1 > hp[v]) hp[v] = d1;
            } else {
                if (d1 < hn[v]) hn[v] = d1;
            }
        }

        __syncthreads();   // all waves done reading buf[cur] before it is rewritten
    }

    // Reduce across the 16 lanes of each half (xor masks <=8 stay in a half).
#pragma unroll
    for (int v = 0; v < 8; ++v) {
        float mp = hp[v], mn = hn[v];
#pragma unroll
        for (int off = 8; off; off >>= 1) {
            mp = fmaxf(mp, __shfl_xor(mp, off));
            mn = fminf(mn, __shfl_xor(mn, off));
        }
        if (l15 == 0) {
            hpOut[i0 + v + 8 * hi] = mp;
            hnOut[i0 + v + 8 * hi] = mn;
        }
    }
}

// ---------------------------------------------------------------- kernel 2
__global__ __launch_bounds__(256) void triplet_finalize(const float* __restrict__ hp,
                                                        const float* __restrict__ hn,
                                                        float* __restrict__ out) {
    __shared__ float ssum[256];
    __shared__ int   scnt[256];
    float s = 0.0f;
    int   c = 0;
    for (int i = threadIdx.x; i < N_FEATS; i += 256) {
        float tl = hp[i] - hn[i] + MARGIN;
        if (tl > 0.0f) { s += tl; ++c; }
    }
    ssum[threadIdx.x] = s;
    scnt[threadIdx.x] = c;
    __syncthreads();
    for (int off = 128; off; off >>= 1) {
        if ((int)threadIdx.x < off) {
            ssum[threadIdx.x] += ssum[threadIdx.x + off];
            scnt[threadIdx.x] += scnt[threadIdx.x + off];
        }
        __syncthreads();
    }
    if (threadIdx.x == 0)
        out[0] = (scnt[0] > 0) ? (ssum[0] / (float)scnt[0]) : 0.0f;
}

// ---------------------------------------------------------------- launcher
extern "C" void kernel_launch(void* const* d_in, const int* in_sizes, int n_in,
                              void* d_out, int out_size, void* d_ws, size_t ws_size,
                              hipStream_t stream) {
    const float* feats  = (const float*)d_in[0];   // [4096, 512] f32
    const int*   labels = (const int*)d_in[1];     // [4096] i32
    float*       out    = (float*)d_out;           // scalar f32

    // workspace layout: f16 feats (4 MB) | hp (16 KB) | hn (16 KB)
    _Float16* f16 = (_Float16*)d_ws;
    float*    hp  = (float*)((char*)d_ws + (size_t)N_FEATS * DIM * sizeof(_Float16));
    float*    hn  = hp + N_FEATS;

    const int nelem = N_FEATS * DIM;
    cvt_f32_to_f16<<<(nelem + 255) / 256, 256, 0, stream>>>(feats, f16, nelem);

    triplet_mine<<<N_FEATS / 16 / 8, 256, 0, stream>>>(f16, labels, hp, hn);

    triplet_finalize<<<1, 256, 0, stream>>>(hp, hn, out);
}
